// DeconvDecoder_58720792871211
// MI455X (gfx1250) — matrix operand order repaired and checked by reading, not compiled, not run on hardware
//
#include <hip/hip_runtime.h>

// ============================================================================
// DeconvDecoder for MI455X (gfx1250, wave32, WMMA).
// GEMM-shaped work (dense deconvs, 1x1 conv, switched expert deconvs) runs
// through v_wmma_f32_16x16x32_bf16 (bf16 operands, f32 accumulate).
// Bandwidth-bound stages (FC chain, depthwise deconv) are VALU kernels.
// Sample tiles are staged into LDS with gfx1250 async global->LDS copies
// (ASYNCcnt) where available.
// ============================================================================

typedef __bf16 bf16_t;
typedef __attribute__((ext_vector_type(16))) __bf16 v16bf;
typedef __attribute__((ext_vector_type(8)))  float  v8f;

static constexpr int BATCH = 2048;

#ifndef __has_builtin
#define __has_builtin(x) 0
#endif
#if __has_builtin(__builtin_amdgcn_global_load_async_to_lds_b128)
#define HAVE_ASYNC_LDS 1
// Builtin signature (from compiler diagnostic): param0 is
// '__device__' (addrspace(1)) pointer to int __attribute__((vector_size(16))).
typedef int vint4 __attribute__((vector_size(4 * sizeof(int))));
typedef __attribute__((address_space(1))) vint4 as1_vint4;
typedef __attribute__((address_space(3))) vint4 as3_vint4;
#else
#define HAVE_ASYNC_LDS 0
#endif

__device__ __forceinline__ int iclampi(int v, int lo, int hi) {
  return v < lo ? lo : (v > hi ? hi : v);
}
__device__ __forceinline__ float relu(float v) { return v > 0.0f ? v : 0.0f; }

// ---------------------------------------------------------------------------
// Kernel 1: out[b,d] = sum_k z2[b,k] * W[k,d] + bias[d]   (K=10, bandwidth)
// ---------------------------------------------------------------------------
__global__ __launch_bounds__(256)
void fc_latent_kernel(const float* __restrict__ z2, const float* __restrict__ w,
                      const float* __restrict__ bias, float* __restrict__ out)
{
  const int idx = blockIdx.x * 256 + threadIdx.x;   // b*1024 + d
  const int b = idx >> 10;
  const int d = idx & 1023;
  float acc = bias[d];
  #pragma unroll
  for (int k = 0; k < 10; ++k) acc += z2[b * 10 + k] * w[k * 1024 + d];
  out[idx] = acc;
}

// ---------------------------------------------------------------------------
// Kernel 2: switched FC (rank-6 bottleneck, per-sample expert).
// One block per sample; 256 threads * 4 elems = D=1024.
// ---------------------------------------------------------------------------
__global__ __launch_bounds__(256)
void switched_fc_kernel(const float* __restrict__ x, float* __restrict__ out,
                        const int* __restrict__ ys, const float* __restrict__ zs,
                        const float* __restrict__ W1, const float* __restrict__ b1,
                        const float* __restrict__ W2, const float* __restrict__ b2)
{
  constexpr int D = 1024, M = 6;
  __shared__ float red[M][256];

  const int b   = blockIdx.x;
  const int tid = threadIdx.x;
  const int ex  = ys[b];
  const float z = zs[b];

  const float4 xv = ((const float4*)(x + (size_t)b * D))[tid];
  const int d0 = tid * 4;

  float acc[M];
  #pragma unroll
  for (int m = 0; m < M; ++m) acc[m] = 0.0f;
  #pragma unroll
  for (int q = 0; q < 4; ++q) {
    const float h = relu((&xv.x)[q]);
    const float* wr = W1 + ((size_t)ex * D + d0 + q) * M;
    #pragma unroll
    for (int m = 0; m < M; ++m) acc[m] += h * wr[m];
  }
  #pragma unroll
  for (int m = 0; m < M; ++m) red[m][tid] = acc[m];
  __syncthreads();
  for (int s = 128; s > 0; s >>= 1) {
    if (tid < s) {
      #pragma unroll
      for (int m = 0; m < M; ++m) red[m][tid] += red[m][tid + s];
    }
    __syncthreads();
  }
  float t[M];
  #pragma unroll
  for (int m = 0; m < M; ++m) t[m] = relu(red[m][0] + b1[ex * M + m]);

  float4 o;
  #pragma unroll
  for (int q = 0; q < 4; ++q) {
    const int d = d0 + q;
    float s2 = b2[(size_t)ex * D + d];
    #pragma unroll
    for (int m = 0; m < M; ++m) s2 += t[m] * W2[((size_t)ex * M + m) * D + d];
    (&o.x)[q] = (&xv.x)[q] + s2 * z;
  }
  ((float4*)(out + (size_t)b * D))[tid] = o;
}

// ---------------------------------------------------------------------------
// Kernel 3: generic dense ConvTranspose2d as parity-class implicit GEMM.
//   out[b,co,y,x] = bias[co] + sum_{ci,r,s : y=STR*i-PAD+r, x=STR*j-PAD+s}
//                   in[b,ci,i,j] * w[ci,co,r,s]
// Per output parity class exactly TD=KS/STR taps/dim are live -> K = CI*TD^2
// with zero waste. One wave per 16-channel N-tile; A staged in LDS (bf16)
// shared by all waves; B staged col-major in LDS once per class.
// Staging is branch-free: clamped unconditional loads + select.
// ---------------------------------------------------------------------------
template<int CI, int CO, int HI, int WI, int HO, int WO,
         int KS, int STR, int PAD, bool INRELU, bool OUTRELU>
__global__ __launch_bounds__(32 * ((CO + 15) / 16))
void deconv_wmma_kernel(const float* __restrict__ in, const float* __restrict__ w,
                        const float* __restrict__ bias, float* __restrict__ out)
{
  constexpr int TD    = KS / STR;
  constexpr int TAPS  = TD * TD;
  constexpr int K     = CI * TAPS;
  constexpr int NT    = (CO + 15) / 16;
  constexpr int NTHR  = 32 * NT;
  constexpr int CHO   = HO / STR;
  constexpr int CWO   = WO / STR;
  constexpr int MT    = (CHO * CWO) / 16;
  constexpr int KSTEP = K / 32;
  static_assert(K % 32 == 0, "K must be a multiple of 32");
  static_assert((CHO * CWO) % 16 == 0, "class pixels must tile by 16");

  __shared__ bf16_t As[16][K];          // A tile, row-major [m][k]
  __shared__ bf16_t Bs[NT][16][K];      // B tiles, col-major [n][k]

  const int b   = blockIdx.x;           // sample
  const int cls = blockIdx.y;           // parity class
  const int py  = cls / STR;
  const int px  = cls % STR;

  const int tid  = threadIdx.x;
  const int wave = tid >> 5;
  const int lane = tid & 31;
  const int half = lane >> 4;
  const int nl   = lane & 15;

  // ---- stage weights for this parity class (col-major, bf16) ----
  const int r0 = (py + PAD) % STR;
  const int s0 = (px + PAD) % STR;
  for (int idx = tid; idx < NT * 16 * K; idx += NTHR) {
    const int k  = idx % K;
    const int n  = (idx / K) % 16;
    const int nt = idx / (16 * K);
    const int co = iclampi(nt * 16 + n, 0, CO - 1);
    const int ci = k / TAPS;
    const int t  = k % TAPS;
    const int r  = r0 + (t / TD) * STR;
    const int s  = s0 + (t % TD) * STR;
    const float val = w[((ci * CO + co) * KS + r) * KS + s];
    Bs[nt][n][k] = (nt * 16 + n < CO) ? (bf16_t)val : (bf16_t)0.0f;
  }

  const float* inb = in + (size_t)b * CI * HI * WI;

  for (int mt = 0; mt < MT; ++mt) {
    __syncthreads();
    // ---- stage A: 16 output pixels of this class, gather taps (zero-pad) ----
    for (int idx = tid; idx < 16 * K; idx += NTHR) {
      const int k   = idx % K;
      const int row = idx / K;
      const int p   = mt * 16 + row;
      const int y   = (p / CWO) * STR + py;
      const int x   = (p % CWO) * STR + px;
      const int ci  = k / TAPS;
      const int t   = k % TAPS;
      const int i   = (y + PAD) / STR - (t / TD);
      const int j   = (x + PAD) / STR - (t % TD);
      const bool ok = ((unsigned)i < (unsigned)HI) && ((unsigned)j < (unsigned)WI);
      float val = inb[(ci * HI + iclampi(i, 0, HI - 1)) * WI + iclampi(j, 0, WI - 1)];
      if (INRELU) val = relu(val);
      As[row][k] = ok ? (bf16_t)val : (bf16_t)0.0f;
    }
    __syncthreads();

    // ---- WMMA: A(16xK) * B(Kx16) per wave's N-tile ----
    v8f acc = {0.f, 0.f, 0.f, 0.f, 0.f, 0.f, 0.f, 0.f};
    #pragma unroll
    for (int ks = 0; ks < KSTEP; ++ks) {
      v16bf afrag, bfrag;
      // A frag (ISA 16-bit A 16x32): lane holds row nl; K = half*8+e, 16+half*8+e
      const bf16_t* ap = &As[nl][ks * 32 + half * 8];
      #pragma unroll
      for (int e = 0; e < 8; ++e) { afrag[e] = ap[e]; afrag[e + 8] = ap[16 + e]; }
      // B frag: lane holds col nl; K = half*8+e, 16+half*8+e (matches A)
      const bf16_t* bp = &Bs[wave][nl][ks * 32 + half * 8];
      #pragma unroll
      for (int e = 0; e < 8; ++e) { bfrag[e] = bp[e]; bfrag[e + 8] = bp[16 + e]; }
      acc = __builtin_amdgcn_wmma_f32_16x16x32_bf16(false, afrag, false, bfrag,
                                                    (short)0, acc, false, false);
    }

    // ---- writeout: D elem v -> m = v + 8*half, n = nl ----
    const int co = wave * 16 + nl;
    if (co < CO) {
      const float bb = bias[co];
      #pragma unroll
      for (int vv = 0; vv < 8; ++vv) {
        const int m = vv + half * 8;
        const int p = mt * 16 + m;
        const int y = (p / CWO) * STR + py;
        const int x = (p % CWO) * STR + px;
        float r = acc[vv] + bb;
        if (OUTRELU) r = relu(r);
        out[(((size_t)b * CO + co) * HO + y) * WO + x] = r;
      }
    }
    __syncthreads();
  }
}

// ---------------------------------------------------------------------------
// Kernel 4: depthwise ConvTranspose2d(32,g=32,4,s=2,p=1), 8x8 -> 16x16. VALU.
// Branch-free: clamped loads + mask-select.
// ---------------------------------------------------------------------------
__global__ __launch_bounds__(256)
void dw_deconv_kernel(const float* __restrict__ in, const float* __restrict__ w,
                      const float* __restrict__ bias, float* __restrict__ out)
{
  const int idx = blockIdx.x * 256 + threadIdx.x;   // b*32*256 + c*256 + y*16 + x
  const int xo = idx & 15;
  const int y  = (idx >> 4) & 15;
  const int c  = (idx >> 8) & 31;
  const int b  = idx >> 13;
  float acc = bias[c];
  const int ib = (y + 1) >> 1, jb = (xo + 1) >> 1;
  const float* inc = in + (((size_t)b * 32 + c) * 64);
  #pragma unroll
  for (int qy = 0; qy < 2; ++qy) {
    const int i  = ib - qy;
    const int r  = ((y + 1) & 1) + 2 * qy;
    const bool oky = (unsigned)i < 8u;
    #pragma unroll
    for (int qx = 0; qx < 2; ++qx) {
      const int j  = jb - qx;
      const int s  = ((xo + 1) & 1) + 2 * qx;
      const bool ok = oky && ((unsigned)j < 8u);
      const float v = inc[iclampi(i, 0, 7) * 8 + iclampi(j, 0, 7)];
      acc += (ok ? v : 0.0f) * w[c * 16 + r * 4 + s];
    }
  }
  out[idx] = acc;
}

// ---------------------------------------------------------------------------
// Kernel 5: switched deconv block, one block (128 thr / 4 waves) per sample.
//   h = relu(x); t = relu(deconv2x2(h,W1[e])[:S,:S] + b1[e])
//   out = x + (deconv2x2(t,W2[e])[:S,:S] + b2[e]) * z
// The sample (32x16x16 f32) is async-copied contiguously into LDS (Xs, raw,
// for the residual), then a zero-padded bf16 relu copy (Hs) is built once so
// every stage-1 A-fragment gather is a single unconditional 16-bit LDS load.
// Tap decode uses qx = 1-(t&1) so consecutive fragment elements read
// ascending LDS addresses (merged loads need no half-swaps); the B-weight
// staging uses the identical flipped decode, so the contraction is unchanged.
// Stage 1: N=6 (padded to 16), K=32*4=128.
// Stage 2: N=32 (2 tiles),     K=6*4=24 padded to 32 (zeroed planes 6,7).
// ---------------------------------------------------------------------------
__global__ __launch_bounds__(128)
void switched_deconv_kernel(const float* __restrict__ x, float* __restrict__ out,
                            const int* __restrict__ ys, const float* __restrict__ zs,
                            const float* __restrict__ W1, const float* __restrict__ b1,
                            const float* __restrict__ W2, const float* __restrict__ b2)
{
  constexpr int CH = 32, CM = 6, S = 16, SP = S + 1;   // 17: zero top row/left col
  constexpr int K1 = CH * 4;                           // 128
  __shared__ float  Xs[CH * S * S];                    // raw x, contiguous (32 KB)
  __shared__ bf16_t Hs[CH][SP][SP];                    // relu(x) bf16, padded (18 KB)
  __shared__ bf16_t B1s[16][K1];                       // stage-1 weights, col-major
  __shared__ bf16_t B2s[2][16][32];                    // stage-2 weights, col-major
  __shared__ bf16_t T1p[8][SP][SP];                    // mid acts, planes 6,7 zero

  const int b  = blockIdx.x;
  const int ex = ys[b];
  const float z = zs[b];
  const int tid  = threadIdx.x;
  const int wave = tid >> 5;
  const int lane = tid & 31;
  const int half = lane >> 4;
  const int nl   = lane & 15;

  const float* xb = x   + (size_t)b * CH * S * S;
  float*       ob = out + (size_t)b * CH * S * S;

  // ---- async-copy the raw sample into LDS (contiguous b128 chunks) ----
#if HAVE_ASYNC_LDS
  for (int idx = tid; idx < CH * S * S / 4; idx += 128) {
    const float* g = xb + idx * 4;
    float* l = &Xs[idx * 4];
    __builtin_amdgcn_global_load_async_to_lds_b128(
        (as1_vint4*)(void*)const_cast<float*>(g),
        (as3_vint4*)(void*)l, 0, 0);
  }
#else
  for (int idx = tid; idx < CH * S * S; idx += 128) Xs[idx] = xb[idx];
#endif

  // ---- zero T1p entirely (borders + K-padding planes) ----
  for (int idx = tid; idx < 8 * SP * SP; idx += 128)
    ((bf16_t*)T1p)[idx] = (bf16_t)0.0f;
  // ---- zero Hs borders (row 0 and col 0 of each plane) ----
  for (int idx = tid; idx < CH * (SP + S); idx += 128) {
    const int c = idx / (SP + S);
    const int r = idx % (SP + S);
    if (r < SP) Hs[c][0][r] = (bf16_t)0.0f;
    else        Hs[c][r - S][0] = (bf16_t)0.0f;   // rows 1..16, col 0
  }
  // ---- stage expert weights into LDS (bf16, col-major, zero-padded) ----
  // Tap decode (flipped qx): t -> (r = t>>1, s = 1-(t&1)).
  for (int idx = tid; idx < 16 * K1; idx += 128) {
    const int k = idx % K1, n = idx / K1;
    const int ci = k >> 2, t = k & 3;
    const int nn = iclampi(n, 0, CM - 1);
    const float v =
        W1[(((ex * CH + ci) * CM + nn) * 2 + (t >> 1)) * 2 + (1 - (t & 1))];
    B1s[n][k] = (n < CM) ? (bf16_t)v : (bf16_t)0.0f;
  }
  for (int idx = tid; idx < 2 * 16 * 32; idx += 128) {
    const int k = idx & 31, n = (idx >> 5) & 15, nt = idx >> 9;
    const int ci = iclampi(k >> 2, 0, CM - 1), t = k & 3;
    const int co = nt * 16 + n;
    const float v =
        W2[(((ex * CM + ci) * CH + co) * 2 + (t >> 1)) * 2 + (1 - (t & 1))];
    B2s[nt][n][k] = ((k >> 2) < CM) ? (bf16_t)v : (bf16_t)0.0f;
  }
#if HAVE_ASYNC_LDS
#if __has_builtin(__builtin_amdgcn_s_wait_asynccnt)
  __builtin_amdgcn_s_wait_asynccnt(0);
#else
  asm volatile("s_wait_asynccnt 0x0" ::: "memory");
#endif
#endif
  __syncthreads();

  // ---- build zero-padded bf16 relu copy once (amortizes relu+cvt 4x) ----
  for (int idx = tid; idx < CH * S * S; idx += 128) {
    const int c = idx >> 8, i = (idx >> 4) & 15, j = idx & 15;
    Hs[c][i + 1][j + 1] = (bf16_t)relu(Xs[idx]);
  }
  __syncthreads();

  // ---- stage 1: rows y of the 16x16 image; wave w takes y = w, w+4, ... ----
  // A element for tap t: i = y - (t>>1), j = nl - 1 + (t&1)  (ascending pairs)
  for (int y = wave; y < S; y += 4) {
    v8f acc = {0.f, 0.f, 0.f, 0.f, 0.f, 0.f, 0.f, 0.f};
    #pragma unroll
    for (int ks = 0; ks < K1 / 32; ++ks) {
      const int kb = ks * 32;
      v16bf afrag, bfrag;
      #pragma unroll
      for (int e = 0; e < 8; ++e) {
        {
          const int k = kb + half * 8 + e;
          const int t = k & 3;
          afrag[e] = Hs[k >> 2][y - (t >> 1) + 1][nl + (t & 1)];
        }
        {
          const int k = kb + 16 + half * 8 + e;
          const int t = k & 3;
          afrag[e + 8] = Hs[k >> 2][y - (t >> 1) + 1][nl + (t & 1)];
        }
      }
      const bf16_t* bp = &B1s[nl][kb + half * 8];
      #pragma unroll
      for (int e = 0; e < 8; ++e) { bfrag[e] = bp[e]; bfrag[e + 8] = bp[16 + e]; }
      acc = __builtin_amdgcn_wmma_f32_16x16x32_bf16(false, afrag, false, bfrag,
                                                    (short)0, acc, false, false);
    }
    if (nl < CM) {
      const float bb = b1[ex * CM + nl];
      #pragma unroll
      for (int vv = 0; vv < 8; ++vv) {
        const int xo = vv + half * 8;
        T1p[nl][y + 1][xo + 1] = (bf16_t)relu(acc[vv] + bb);
      }
    }
  }
  __syncthreads();

  // ---- stage 2: 16 rows x 2 N-tiles = 32 jobs; wave w takes w, w+4, ... ----
  for (int job = wave; job < 32; job += 4) {
    const int y  = job >> 1;
    const int nt = job & 1;
    v16bf afrag, bfrag;
    #pragma unroll
    for (int e = 0; e < 8; ++e) {
      {
        const int k = half * 8 + e;         // planes 6,7 of T1p are zero => K pad
        const int t = k & 3;
        afrag[e] = T1p[k >> 2][y - (t >> 1) + 1][nl + (t & 1)];
      }
      {
        const int k = 16 + half * 8 + e;
        const int t = k & 3;
        afrag[e + 8] = T1p[k >> 2][y - (t >> 1) + 1][nl + (t & 1)];
      }
    }
    const bf16_t* bp = &B2s[nt][nl][half * 8];
    #pragma unroll
    for (int e = 0; e < 8; ++e) { bfrag[e] = bp[e]; bfrag[e + 8] = bp[16 + e]; }
    v8f acc = {0.f, 0.f, 0.f, 0.f, 0.f, 0.f, 0.f, 0.f};
    acc = __builtin_amdgcn_wmma_f32_16x16x32_bf16(false, afrag, false, bfrag,
                                                  (short)0, acc, false, false);
    const int co = nt * 16 + nl;
    const float bb = b2[ex * CH + co];
    #pragma unroll
    for (int vv = 0; vv < 8; ++vv) {
      const int xo = vv + half * 8;
      const float res = Xs[(co * S + y) * S + xo] + (acc[vv] + bb) * z;
      ob[(co * S + y) * S + xo] = res;
    }
  }
}

// ===========================================================================
// Host-side launcher
// ===========================================================================
extern "C" void kernel_launch(void* const* d_in, const int* in_sizes, int n_in,
                              void* d_out, int out_size, void* d_ws, size_t ws_size,
                              hipStream_t stream)
{
  (void)in_sizes; (void)n_in; (void)out_size; (void)ws_size;

  const float* z2      = (const float*)d_in[0];
  const int*   ys      = (const int*)  d_in[1];
  const float* zs      = (const float*)d_in[2];
  const float* fcw     = (const float*)d_in[3];
  const float* fcb     = (const float*)d_in[4];
  const float* fcsw_w1 = (const float*)d_in[5];
  const float* fcsw_b1 = (const float*)d_in[6];
  const float* fcsw_w2 = (const float*)d_in[7];
  const float* fcsw_b2 = (const float*)d_in[8];
  const float* dcsw_w1 = (const float*)d_in[9];
  const float* dcsw_b1 = (const float*)d_in[10];
  const float* dcsw_w2 = (const float*)d_in[11];
  const float* dcsw_b2 = (const float*)d_in[12];
  const float* w_d1    = (const float*)d_in[13];
  const float* b_d1    = (const float*)d_in[14];
  const float* w_d2a   = (const float*)d_in[15];
  const float* b_d2a   = (const float*)d_in[16];
  const float* w_d2b   = (const float*)d_in[17];
  const float* b_d2b   = (const float*)d_in[18];
  const float* w_d3    = (const float*)d_in[19];
  const float* b_d3    = (const float*)d_in[20];
  const float* w_d4    = (const float*)d_in[21];
  const float* b_d4    = (const float*)d_in[22];
  float* outp = (float*)d_out;

  float* ws = (float*)d_ws;
  size_t off = 0;
  auto carve = [&](size_t nfloats) {
    float* p = ws + off;
    off += (nfloats + 63) & ~(size_t)63;   // 256B aligned slabs
    return p;
  };
  float* X0 = carve((size_t)BATCH * 1024);            // fc ping
  float* X1 = carve((size_t)BATCH * 1024);            // fc pong
  float* C1 = carve((size_t)BATCH * 64 * 8 * 8);      // d1 out
  float* C2 = carve((size_t)BATCH * 32 * 8 * 8);      // d2a out
  float* P0 = carve((size_t)BATCH * 32 * 16 * 16);    // d2b out / switched ping
  float* P1 = carve((size_t)BATCH * 32 * 16 * 16);    // switched pong
  float* C5 = carve((size_t)BATCH * 32 * 32 * 32);    // d3 out

  // --- latent FC ---
  fc_latent_kernel<<<(BATCH * 1024) / 256, 256, 0, stream>>>(z2, fcw, fcb, X0);

  // --- 6 switched FC layers (layer j uses conditioning index 9-j) ---
  float* fin = X0; float* fout = X1;
  for (int j = 0; j < 6; ++j) {
    const int i = 9 - j;
    switched_fc_kernel<<<BATCH, 256, 0, stream>>>(
        fin, fout, ys + (size_t)i * BATCH, zs + (size_t)i * BATCH,
        fcsw_w1 + (size_t)j * 3 * 1024 * 6, fcsw_b1 + (size_t)j * 3 * 6,
        fcsw_w2 + (size_t)j * 3 * 6 * 1024, fcsw_b2 + (size_t)j * 3 * 1024);
    float* t = fin; fin = fout; fout = t;
  }
  // fin == X0 after 6 swaps

  // --- d1: relu(in) -> ConvT(64,64,4,s2,p1) -> relu, 4x4 -> 8x8 ---
  deconv_wmma_kernel<64, 64, 4, 4, 8, 8, 4, 2, 1, true, true>
      <<<dim3(BATCH, 4), 128, 0, stream>>>(fin, w_d1, b_d1, C1);

  // --- d2a: 1x1 ConvT(64,32), 8x8 ---
  deconv_wmma_kernel<64, 32, 8, 8, 8, 8, 1, 1, 0, false, false>
      <<<dim3(BATCH, 1), 64, 0, stream>>>(C1, w_d2a, b_d2a, C2);

  // --- d2b: depthwise ConvT(32,g=32,4,s2,p1), 8x8 -> 16x16 ---
  dw_deconv_kernel<<<(BATCH * 32 * 16 * 16) / 256, 256, 0, stream>>>(C2, w_d2b, b_d2b, P0);

  // --- 4 switched deconv layers (layer j uses conditioning index 3-j) ---
  float* pin = P0; float* pout = P1;
  for (int j = 0; j < 4; ++j) {
    const int i = 3 - j;
    switched_deconv_kernel<<<BATCH, 128, 0, stream>>>(
        pin, pout, ys + (size_t)i * BATCH, zs + (size_t)i * BATCH,
        dcsw_w1 + (size_t)j * 3 * 32 * 6 * 4, dcsw_b1 + (size_t)j * 3 * 6,
        dcsw_w2 + (size_t)j * 3 * 6 * 32 * 4, dcsw_b2 + (size_t)j * 3 * 32);
    float* t = pin; pin = pout; pout = t;
  }
  // pin == P0 after 4 swaps

  // --- d3: relu(in) -> ConvT(32,32,4,s2,p1) -> relu, 16x16 -> 32x32 ---
  deconv_wmma_kernel<32, 32, 16, 16, 32, 32, 4, 2, 1, true, true>
      <<<dim3(BATCH, 4), 64, 0, stream>>>(pin, w_d3, b_d3, C5);

  // --- d4: ConvT(32,3,4,s2,p1), 32x32 -> 64x64, no relu ---
  deconv_wmma_kernel<32, 3, 32, 32, 64, 64, 4, 2, 1, false, false>
      <<<dim3(BATCH, 4), 32, 0, stream>>>(C5, w_d4, b_d4, outp);
}
